// MambaBlock_56942676411182
// MI455X (gfx1250) — matrix operand order repaired and checked by reading, not compiled, not run on hardware
//
#include <hip/hip_runtime.h>

#define B_SZ      2
#define L_SZ      2048
#define DMODEL    1024
#define DSTATE    128
#define DCONV     4
#define DINNER    2048
#define NHEADS    32
#define HEADDIM   64
#define DINPROJ   4384          // 2*DINNER + 2*DSTATE + NHEADS
#define CONVDIM   2304          // DINNER + 2*DSTATE
#define NTOK      (B_SZ * L_SZ) // 4096
#define DFF       4096

typedef __attribute__((ext_vector_type(16))) __bf16 v16bf;
typedef __attribute__((ext_vector_type(8)))  float  v8f;

union BfFrag { uint4 u[2]; v16bf v; };
struct FragSet { BfFrag a0, a1, b0, b1; };

__device__ __forceinline__ unsigned short f2bf(float f) {
  unsigned u = __float_as_uint(f);
  u += 0x7FFFu + ((u >> 16) & 1u);           // round-to-nearest-even
  return (unsigned short)(u >> 16);
}
__device__ __forceinline__ float siluf(float x) { return x / (1.f + expf(-x)); }

__device__ __forceinline__ float block_reduce_sum_256(float v) {
  __shared__ float wsum[8];
  #pragma unroll
  for (int off = 16; off > 0; off >>= 1) v += __shfl_down(v, off, 32);
  if ((threadIdx.x & 31) == 0) wsum[threadIdx.x >> 5] = v;
  __syncthreads();
  float t = 0.f;
  #pragma unroll
  for (int i = 0; i < 8; ++i) t += wsum[i];
  return t;
}

// ---------------- f32 -> bf16 convert (weights, once per launch) ----------------
__global__ __launch_bounds__(256)
void cvt_f32_bf16(const float* __restrict__ in, unsigned short* __restrict__ out, int n) {
  int i = blockIdx.x * 256 + threadIdx.x;
  if (i < n) out[i] = f2bf(in[i]);
}

// ---------------- _block_norm: w * x * rsqrt(sum(x^2) + eps), bf16 out ----------------
__global__ __launch_bounds__(256)
void rmsnorm_sum_bf16(const float* __restrict__ x, const float* __restrict__ w,
                      unsigned short* __restrict__ out, int dim, float eps) {
  const int t = blockIdx.x;
  const float* xr = x + (size_t)t * dim;
  float ss = 0.f;
  for (int i = threadIdx.x; i < dim; i += 256) { float v = xr[i]; ss += v * v; }
  float inv = rsqrtf(block_reduce_sum_256(ss) + eps);
  for (int i = threadIdx.x; i < dim; i += 256)
    out[(size_t)t * dim + i] = f2bf(w[i] * xr[i] * inv);
}

// ---------------- WMMA GEMM helpers ----------------
__device__ __forceinline__ void load_frags(FragSet& f,
                                           const unsigned short* __restrict__ A0,
                                           const unsigned short* __restrict__ A1,
                                           const unsigned short* __restrict__ B0,
                                           const unsigned short* __restrict__ B1, int k) {
  f.a0.u[0] = *(const uint4*)(A0 + k);  f.a0.u[1] = *(const uint4*)(A0 + k + 8);
  f.a1.u[0] = *(const uint4*)(A1 + k);  f.a1.u[1] = *(const uint4*)(A1 + k + 8);
  f.b0.u[0] = *(const uint4*)(B0 + k);  f.b0.u[1] = *(const uint4*)(B0 + k + 8);
  f.b1.u[0] = *(const uint4*)(B1 + k);  f.b1.u[1] = *(const uint4*)(B1 + k + 8);
}

__device__ __forceinline__ void mma4(const FragSet& f, v8f& c00, v8f& c01,
                                     v8f& c10, v8f& c11) {
  c00 = __builtin_amdgcn_wmma_f32_16x16x32_bf16(false, f.a0.v, false, f.b0.v,
                                                (short)0, c00, false, false);
  c01 = __builtin_amdgcn_wmma_f32_16x16x32_bf16(false, f.a0.v, false, f.b1.v,
                                                (short)0, c01, false, false);
  c10 = __builtin_amdgcn_wmma_f32_16x16x32_bf16(false, f.a1.v, false, f.b0.v,
                                                (short)0, c10, false, false);
  c11 = __builtin_amdgcn_wmma_f32_16x16x32_bf16(false, f.a1.v, false, f.b1.v,
                                                (short)0, c11, false, false);
}

// ---------------- WMMA GEMM: C(MxN,f32) = A(MxK,bf16) @ B(NxK,bf16)^T ----------------
// Wave tile 32x32 (4 accumulators). Ping-pong pipelined K-loop (2 chunks / iter):
// fixed fragment register sets P/Q alternate, so no rotation copies; each 4-WMMA
// group consumes fragments whose loads were issued a half-iteration earlier.
// Block = 8 waves -> 256x32. Requires: K % 64 == 0, M % 256 == 0, N % 32 == 0.
__global__ __launch_bounds__(256)
void gemm_bf16_wmma(const unsigned short* __restrict__ A,
                    const unsigned short* __restrict__ B,
                    float* __restrict__ C, int M, int N, int K) {
  const int lane = threadIdx.x & 31;
  const int wave = threadIdx.x >> 5;
  const int ln   = lane & 15;
  const int hf   = lane >> 4;
  const int m0 = blockIdx.y * 256 + wave * 32;
  const int n0 = blockIdx.x * 32;

  // Per-lane fragment base pointers (per-lane K offsets folded in).
  const unsigned short* A0 = A + (size_t)(m0 + ln) * K + 16 * hf;       // A rows m0..m0+15
  const unsigned short* A1 = A + (size_t)(m0 + 16 + ln) * K + 16 * hf;  // A rows m0+16..m0+31
  const unsigned short* B0 = B + (size_t)(n0 + ln) * K + 16 * hf;       // W rows n0..n0+15
  const unsigned short* B1 = B + (size_t)(n0 + 16 + ln) * K + 16 * hf;  // W rows n0+16..n0+31

  v8f c00 = {}, c01 = {}, c10 = {}, c11 = {};

  FragSet P, Q;
  load_frags(P, A0, A1, B0, B1, 0);

  for (int k = 0; k + 64 <= K; k += 64) {
    // chunk k+32 loads in flight while chunk k computes
    load_frags(Q, A0, A1, B0, B1, k + 32);
    __builtin_prefetch(A0 + k + 128, 0, 1);   // gfx1250 global_prefetch_b8
    __builtin_prefetch(A1 + k + 128, 0, 1);
    __builtin_prefetch(B0 + k + 128, 0, 1);
    __builtin_prefetch(B1 + k + 128, 0, 1);
    mma4(P, c00, c01, c10, c11);
    if (k + 64 < K) {
      load_frags(P, A0, A1, B0, B1, k + 64);
      __builtin_prefetch(A0 + k + 160, 0, 1);
      __builtin_prefetch(B0 + k + 160, 0, 1);
    }
    mma4(Q, c00, c01, c10, c11);
  }

  // C/D layout: lane col = ln, VGPR r -> row r + 8*hf.
  float* Cp0 = C + (size_t)(m0 + 8 * hf) * N + n0 + ln;
  float* Cp1 = C + (size_t)(m0 + 16 + 8 * hf) * N + n0 + ln;
  #pragma unroll
  for (int r = 0; r < 8; ++r) {
    Cp0[(size_t)r * N]      = c00[r];
    Cp0[(size_t)r * N + 16] = c01[r];
    Cp1[(size_t)r * N]      = c10[r];
    Cp1[(size_t)r * N + 16] = c11[r];
  }
}

// ---------------- causal depthwise conv1d + silu over xBC ----------------
__global__ __launch_bounds__(256)
void conv_silu(const float* __restrict__ zxbcdt, const float* __restrict__ cw,
               const float* __restrict__ cb, float* __restrict__ xbc) {
  int idx = blockIdx.x * 256 + threadIdx.x;
  if (idx >= NTOK * CONVDIM) return;
  int c = idx % CONVDIM;
  int l = (idx / CONVDIM) % L_SZ;
  int b = idx / (CONVDIM * L_SZ);
  float acc = cb[c];
  #pragma unroll
  for (int k = 0; k < DCONV; ++k) {
    int ls = l - (DCONV - 1) + k;
    float v = (ls >= 0) ? zxbcdt[(size_t)(b * L_SZ + ls) * DINPROJ + DINNER + c] : 0.f;
    acc += v * cw[c * DCONV + k];
  }
  xbc[(size_t)(b * L_SZ + l) * CONVDIM + c] = siluf(acc);
}

// ---------------- dt = softplus(dt_raw + dt_bias) ----------------
__global__ __launch_bounds__(256)
void dt_softplus(const float* __restrict__ zxbcdt, const float* __restrict__ dt_bias,
                 float* __restrict__ dtb) {
  int i = blockIdx.x * 256 + threadIdx.x;
  if (i >= NTOK * NHEADS) return;
  int h = i & (NHEADS - 1);
  int tok = i >> 5;
  float v = zxbcdt[(size_t)tok * DINPROJ + (2 * DINNER + 2 * DSTATE) + h] + dt_bias[h];
  dtb[i] = (v > 20.f) ? v : log1pf(expf(v));
}

// ---------------- sequential SSM scan: one workgroup per (b,h) ----------------
// 256 threads: p = tid&63, n-chunk = tid>>6 (32 n-values in registers per thread)
__global__ __launch_bounds__(256)
void ssm_scan(const float* __restrict__ xbc, const float* __restrict__ dtb,
              const float* __restrict__ A_log, const float* __restrict__ Dp,
              float* __restrict__ y) {
  const int bh = blockIdx.x;
  const int b = bh >> 5;            // NHEADS = 32
  const int h = bh & 31;
  const int p = threadIdx.x & 63;
  const int nb = threadIdx.x >> 6;
  const int n0 = nb * 32;
  const float Ah = -expf(A_log[h]);
  const float Dh = Dp[h];
  float s[32];
  #pragma unroll
  for (int i = 0; i < 32; ++i) s[i] = 0.f;

  __shared__ float sB[DSTATE], sC[DSTATE], sx[HEADDIM], sred[256];
  __shared__ float sdt;
  for (int l = 0; l < L_SZ; ++l) {
    const size_t tok = (size_t)b * L_SZ + l;
    const float* xr = xbc + tok * CONVDIM;
    if (threadIdx.x < 128)      sB[threadIdx.x]       = xr[DINNER + threadIdx.x];
    else                        sC[threadIdx.x - 128] = xr[DINNER + threadIdx.x];
    if (threadIdx.x < HEADDIM)  sx[threadIdx.x]       = xr[h * HEADDIM + threadIdx.x];
    if (threadIdx.x == 0)       sdt                   = dtb[tok * NHEADS + h];
    __syncthreads();
    const float dt = sdt;
    const float dA = expf(dt * Ah);
    const float coef = dt * sx[p];
    float acc = 0.f;
    #pragma unroll
    for (int i = 0; i < 32; ++i) {
      float sv = s[i] * dA + coef * sB[n0 + i];
      s[i] = sv;
      acc += sv * sC[n0 + i];
    }
    sred[threadIdx.x] = acc;
    __syncthreads();
    if (nb == 0) {
      float yv = sred[p] + sred[p + 64] + sred[p + 128] + sred[p + 192] + Dh * sx[p];
      y[tok * DINNER + h * HEADDIM + p] = yv;
    }
    __syncthreads();
  }
}

// ---------------- y*silu(z), RMS(mean)+1e-5, *mnorm_w -> bf16 ----------------
__global__ __launch_bounds__(256)
void gate_rmsnorm_bf16(const float* __restrict__ y, const float* __restrict__ zxbcdt,
                       const float* __restrict__ w, unsigned short* __restrict__ out) {
  const int t = blockIdx.x;
  const float* yr = y + (size_t)t * DINNER;
  const float* zr = zxbcdt + (size_t)t * DINPROJ;   // z = first DINNER
  float g[8]; float ss = 0.f;
  #pragma unroll
  for (int j = 0; j < 8; ++j) {
    int i = threadIdx.x + j * 256;
    float v = yr[i] * siluf(zr[i]);
    g[j] = v; ss += v * v;
  }
  float inv = rsqrtf(block_reduce_sum_256(ss) * (1.f / DINNER) + 1e-5f);
  #pragma unroll
  for (int j = 0; j < 8; ++j) {
    int i = threadIdx.x + j * 256;
    out[(size_t)t * DINNER + i] = f2bf(g[j] * inv * w[i]);
  }
}

// ---------------- elementwise epilogues ----------------
__global__ __launch_bounds__(256)
void add2(const float* __restrict__ a, const float* __restrict__ b,
          float* __restrict__ o, int n) {
  int i = blockIdx.x * 256 + threadIdx.x;
  if (i < n) o[i] = a[i] + b[i];
}

__global__ __launch_bounds__(256)
void bias_gelu_bf16(const float* __restrict__ hgem, const float* __restrict__ bias,
                    unsigned short* __restrict__ out, int n, int dim) {
  int i = blockIdx.x * 256 + threadIdx.x;
  if (i >= n) return;
  float v = hgem[i] + bias[i % dim];
  out[i] = f2bf(0.5f * v * (1.f + erff(v * 0.70710678118654752f)));
}

__global__ __launch_bounds__(256)
void add2_bias(const float* __restrict__ a, const float* __restrict__ b,
               const float* __restrict__ bias, float* __restrict__ o, int n, int dim) {
  int i = blockIdx.x * 256 + threadIdx.x;
  if (i < n) o[i] = a[i] + b[i] + bias[i % dim];
}

extern "C" void kernel_launch(void* const* d_in, const int* in_sizes, int n_in,
                              void* d_out, int out_size, void* d_ws, size_t ws_size,
                              hipStream_t stream) {
  (void)in_sizes; (void)n_in; (void)out_size; (void)ws_size;
  const float* x         = (const float*)d_in[0];
  const float* norm1_w   = (const float*)d_in[1];
  const float* in_proj_w = (const float*)d_in[2];
  const float* conv_w    = (const float*)d_in[3];
  const float* conv_b    = (const float*)d_in[4];
  const float* dt_bias   = (const float*)d_in[5];
  const float* A_log     = (const float*)d_in[6];
  const float* Dp        = (const float*)d_in[7];
  const float* mnorm_w   = (const float*)d_in[8];
  const float* out_proj_w= (const float*)d_in[9];
  const float* norm2_w   = (const float*)d_in[10];
  const float* mlp_w1    = (const float*)d_in[11];
  const float* mlp_b1    = (const float*)d_in[12];
  const float* mlp_w2    = (const float*)d_in[13];
  const float* mlp_b2    = (const float*)d_in[14];
  float* out = (float*)d_out;

  char* ws = (char*)d_ws;
  size_t off = 0;
  auto alloc = [&](size_t bytes) { size_t o = off; off += (bytes + 255) & ~(size_t)255; return o; };
  unsigned short* wbf_inproj  = (unsigned short*)(ws + alloc(2ull * DINPROJ * DMODEL));
  unsigned short* wbf_outproj = (unsigned short*)(ws + alloc(2ull * DMODEL * DINNER));
  unsigned short* wbf_w1      = (unsigned short*)(ws + alloc(2ull * DFF * DMODEL));
  unsigned short* wbf_w2      = (unsigned short*)(ws + alloc(2ull * DMODEL * DFF));
  unsigned short* bfA         = (unsigned short*)(ws + alloc(2ull * NTOK * DFF));
  float* big    = (float*)(ws + alloc(4ull * NTOK * DINPROJ));   // zxbcdt, later h1
  float* xbc    = (float*)(ws + alloc(4ull * NTOK * CONVDIM));
  float* dtb    = (float*)(ws + alloc(4ull * NTOK * NHEADS));
  float* ybuf   = (float*)(ws + alloc(4ull * NTOK * DINNER));
  float* mamba  = (float*)(ws + alloc(4ull * NTOK * DMODEL));
  float* x2     = (float*)(ws + alloc(4ull * NTOK * DMODEL));
  float* mlpout = (float*)(ws + alloc(4ull * NTOK * DMODEL));

  auto cg = [](long long n) { return dim3((unsigned)((n + 255) / 256)); };

  // weights -> bf16 (deterministic, per launch)
  cvt_f32_bf16<<<cg((long long)DINPROJ * DMODEL), 256, 0, stream>>>(in_proj_w, wbf_inproj, DINPROJ * DMODEL);
  cvt_f32_bf16<<<cg((long long)DMODEL * DINNER),  256, 0, stream>>>(out_proj_w, wbf_outproj, DMODEL * DINNER);
  cvt_f32_bf16<<<cg((long long)DFF * DMODEL),     256, 0, stream>>>(mlp_w1, wbf_w1, DFF * DMODEL);
  cvt_f32_bf16<<<cg((long long)DMODEL * DFF),     256, 0, stream>>>(mlp_w2, wbf_w2, DMODEL * DFF);

  // ---- mamba sublayer ----
  rmsnorm_sum_bf16<<<NTOK, 256, 0, stream>>>(x, norm1_w, bfA, DMODEL, 1e-6f);
  gemm_bf16_wmma<<<dim3(DINPROJ / 32, NTOK / 256), 256, 0, stream>>>(bfA, wbf_inproj, big, NTOK, DINPROJ, DMODEL);
  conv_silu<<<cg((long long)NTOK * CONVDIM), 256, 0, stream>>>(big, conv_w, conv_b, xbc);
  dt_softplus<<<cg((long long)NTOK * NHEADS), 256, 0, stream>>>(big, dt_bias, dtb);
  ssm_scan<<<B_SZ * NHEADS, 256, 0, stream>>>(xbc, dtb, A_log, Dp, ybuf);
  gate_rmsnorm_bf16<<<NTOK, 256, 0, stream>>>(ybuf, big, mnorm_w, bfA);
  gemm_bf16_wmma<<<dim3(DMODEL / 32, NTOK / 256), 256, 0, stream>>>(bfA, wbf_outproj, mamba, NTOK, DMODEL, DINNER);
  add2<<<cg((long long)NTOK * DMODEL), 256, 0, stream>>>(x, mamba, x2, NTOK * DMODEL);

  // ---- MLP sublayer ----
  rmsnorm_sum_bf16<<<NTOK, 256, 0, stream>>>(x2, norm2_w, bfA, DMODEL, 1e-6f);
  gemm_bf16_wmma<<<dim3(DFF / 32, NTOK / 256), 256, 0, stream>>>(bfA, wbf_w1, big, NTOK, DFF, DMODEL);
  bias_gelu_bf16<<<cg((long long)NTOK * DFF), 256, 0, stream>>>(big, mlp_b1, bfA, NTOK * DFF, DFF);
  gemm_bf16_wmma<<<dim3(DMODEL / 32, NTOK / 256), 256, 0, stream>>>(bfA, wbf_w2, mlpout, NTOK, DMODEL, DFF);
  add2_bias<<<cg((long long)NTOK * DMODEL), 256, 0, stream>>>(x2, mlpout, mlp_b2, out, NTOK * DMODEL, DMODEL);
}
